// PrunViTEncoder_38113539785329
// MI455X (gfx1250) — matrix operand order repaired and checked by reading, not compiled, not run on hardware
//
#include <hip/hip_runtime.h>
#include <hip/hip_bf16.h>
#include <math.h>

typedef __attribute__((ext_vector_type(16))) __bf16 v16bf;
typedef __attribute__((ext_vector_type(8)))  float  v8f;

#define L_    12
#define B_    32
#define S_    197
#define D_    768
#define H_    12
#define HD_   64
#define FF_   3072
#define MS_   (B_ * S_)     // 6304 valid rows
#define MROW_ 6400          // row-padded allocation (>= 31*197+256)
#define NB_   16
#define SPK_  208           // kbT row stride  (mult of 8 -> 16B rows)
#define SP2_  224           // scores/probs row stride = padded K (7*32)

// ---------------------------------------------------------------------------
// Batched WMMA GEMM:  out[z] = epilogue(A[z] * B[z])
//   A: bf16 [M,K] row-major (lda), B: bf16 [K,N] row-major (ldb)
//   OUTF -> fp32 C, OUTB -> bf16 Cbf (both ldc); BIAS/RES/GELU in epilogue.
// Block tile 64x128, 8 wave32 waves, wave tile 32x32 = 2x2
// v_wmma_f32_16x16x32_bf16. Staging is UNCONDITIONAL b128 loads (the host
// pads every operand so tile overruns are in-allocation, and K-overrun data
// is zeroed where it could alias valid outputs; all K are multiples of 32),
// register double-buffered over ping-pong LDS so s_wait_loadcnt for tile
// t+1 lands after the WMMAs of tile t. Distance-2 global_prefetch_b8 pulls
// tile t+2 HBM->L2 behind the compute. Stores guarded (row<M, col<N),
// which also discards every product that involved overrun garbage.
// ---------------------------------------------------------------------------
template <bool BIAS, bool RES, bool OUTF, bool OUTB, bool GELU>
__global__ __launch_bounds__(256) void wmma_gemm_bf16(
    const __bf16* __restrict__ A, const __bf16* __restrict__ Bm,
    const float* __restrict__ bias, const float* __restrict__ resid,
    float* __restrict__ C, __bf16* __restrict__ Cbf,
    int M, int N, int K, int lda, int ldb, int ldc,
    long sAb, long sAh, long sBb, long sBh, long sCb, long sCh, int Hdiv)
{
  __shared__ __bf16 As[2][64][40];    // 64x32 tiles (80B rows, 16B aligned)
  __shared__ __bf16 Bs[2][32][136];   // 32x128 tiles (272B rows, 16B aligned)

  const int tid  = threadIdx.x;
  const int lane = tid & 31;
  const int wave = tid >> 5;
  const int wm   = (wave >> 2) * 32;
  const int wn   = (wave & 3) * 32;
  const int bm   = blockIdx.y * 64;
  const int bn   = blockIdx.x * 128;

  const int z  = blockIdx.z;
  const int zb = z / Hdiv;
  const int zh = z % Hdiv;
  A  += zb * sAb + zh * sAh;
  Bm += zb * sBb + zh * sBh;
  const long coff = zb * sCb + zh * sCh;
  if (OUTF) C     += coff;
  if (OUTB) Cbf   += coff;
  if (RES)  resid += coff;

  // staging geometry: 1 A chunk + 2 B chunks (8 bf16 = 16B) per thread
  const int ar  = tid >> 2;            // A row 0..63
  const int ac  = (tid & 3) << 3;      // A col chunk 0/8/16/24
  const int br  = tid >> 4;            // B row 0..15 (and +16)
  const int bc  = (tid & 15) << 3;     // B col chunk 0..120

  auto loadA = [&](int kt) -> uint4 {
    return *(const uint4*)(A + (long)(bm + ar) * lda + (kt + ac));
  };
  auto loadB = [&](int kt, uint4& u0, uint4& u1) {
    u0 = *(const uint4*)(Bm + (long)(kt + br)      * ldb + (bn + bc));
    u1 = *(const uint4*)(Bm + (long)(kt + br + 16) * ldb + (bn + bc));
  };
  auto stash = [&](uint4 a, uint4 b0, uint4 b1, int buf) {
    *(uint4*)&As[buf][ar][ac]      = a;
    *(uint4*)&Bs[buf][br][bc]      = b0;
    *(uint4*)&Bs[buf][br + 16][bc] = b1;
  };

  v8f acc[2][2] = {};

  // A-fragment lane mapping (16-bit A 16x32, ISA 7.12.2):
  //   lanes 0-15: row M=lane,    K = {0..7, 16..23}
  //   lanes16-31: row M=lane-16, K = {8..15, 24..31}
  const int arow = lane & 15;
  const int akb  = (lane >> 4) * 8;

  {
    uint4 a, b0, b1;
    a = loadA(0); loadB(0, b0, b1);
    stash(a, b0, b1, 0);
  }
  __syncthreads();

  const int nk = (K + 31) >> 5;
  for (int t = 0; t < nk; ++t) {
    const int buf = t & 1;
    const bool more = (t + 1 < nk);
    uint4 na, nb0, nb1;
    if (more) { na = loadA((t + 1) << 5); loadB((t + 1) << 5, nb0, nb1); }
    if (t + 2 < nk) {   // distance-2 HBM->L2 prefetch (global_prefetch_b8)
      int kt2 = (t + 2) << 5;
      __builtin_prefetch(A  + (long)(bm + ar) * lda + (kt2 + ac), 0, 1);
      __builtin_prefetch(Bm + (long)(kt2 + br) * ldb + (bn + bc), 0, 1);
    }

    v16bf afr[2], bfr[2];
#pragma unroll
    for (int i = 0; i < 2; ++i) {
#pragma unroll
      for (int e2 = 0; e2 < 16; ++e2) {
        int kk = akb + e2 + ((e2 >= 8) ? 8 : 0);
        afr[i][e2] = As[buf][wm + i * 16 + arow][kk];
      }
    }
    // B fragment: lane l holds column N=l&15, same K set as the A fragment
#pragma unroll
    for (int j = 0; j < 2; ++j) {
#pragma unroll
      for (int e2 = 0; e2 < 16; ++e2)
        bfr[j][e2] = Bs[buf][akb + e2 + ((e2 >= 8) ? 8 : 0)][wn + j * 16 + arow];
    }
#pragma unroll
    for (int i = 0; i < 2; ++i)
#pragma unroll
      for (int j = 0; j < 2; ++j)
        acc[i][j] = __builtin_amdgcn_wmma_f32_16x16x32_bf16(
            false, afr[i], false, bfr[j], (short)0, acc[i][j], false, false);

    if (more) stash(na, nb0, nb1, buf ^ 1);   // s_wait_loadcnt lands here
    __syncthreads();
  }

  // C/D layout: VGPR v -> M = v + (lane>=16 ? 8 : 0), N = lane&15
  const int rsel = (lane >> 4) * 8;
  const int cn   = lane & 15;

  float bv[2] = {0.f, 0.f};
  if (BIAS) {                       // hoisted: 2 loads instead of 32
#pragma unroll
    for (int j = 0; j < 2; ++j) {
      int col = bn + wn + j * 16 + cn;
      bv[j] = (col < N) ? bias[col] : 0.f;
    }
  }
#pragma unroll
  for (int j = 0; j < 2; ++j) {
    const int col = bn + wn + j * 16 + cn;
    if (col < N) {                  // hoisted out of the (i,v) loops
#pragma unroll
      for (int i = 0; i < 2; ++i) {
#pragma unroll
        for (int v = 0; v < 8; ++v) {
          int row = bm + wm + i * 16 + rsel + v;
          if (row < M) {
            float o = acc[i][j][v];
            long idx = (long)row * ldc + col;
            if (BIAS) o += bv[j];
            if (RES)  o += resid[idx];
            if (GELU) o = 0.5f * o * (1.f + erff(o * 0.70710678118654752f));
            if (OUTF) C[idx] = o;
            if (OUTB) Cbf[idx] = (__bf16)o;
          }
        }
      }
    }
  }
}

// ---------------------------------------------------------------------------
// LayerNorm (fp32 in) -> bf16 out, one 256-thread block per row of D
// ---------------------------------------------------------------------------
__global__ __launch_bounds__(256) void layernorm_bf16_k(
    const float* __restrict__ x, const float* __restrict__ w,
    const float* __restrict__ b, __bf16* __restrict__ y, int D)
{
  __shared__ float r1[256], r2[256];
  long row = blockIdx.x;
  const float* xr = x + row * D;
  int tid = threadIdx.x;
  float s = 0.f, s2 = 0.f;
  for (int c = tid; c < D; c += 256) { float v = xr[c]; s += v; s2 += v * v; }
  r1[tid] = s; r2[tid] = s2;
  __syncthreads();
  for (int st = 128; st > 0; st >>= 1) {
    if (tid < st) { r1[tid] += r1[tid + st]; r2[tid] += r2[tid + st]; }
    __syncthreads();
  }
  float mean = r1[0] / (float)D;
  float var  = r2[0] / (float)D - mean * mean;
  float rstd = rsqrtf(var + 1e-12f);
  for (int c = tid; c < D; c += 256)
    y[row * D + c] = (__bf16)((xr[c] - mean) * rstd * w[c] + b[c]);
}

// K^T pre-transpose: kbf[B,S,D] -> kT[(b*H+h)*HD + d][s], row stride SPK
__global__ void ktranspose_bf16_k(const __bf16* __restrict__ src,
                                  __bf16* __restrict__ dst)
{
  long n  = (long)B_ * H_ * HD_ * S_;
  long i  = (long)blockIdx.x * blockDim.x + threadIdx.x;
  long st = (long)gridDim.x * blockDim.x;
  for (; i < n; i += st) {
    int  s  = (int)(i % S_);
    long r  = i / S_;            // (b*H + h)*HD + d
    int  d  = (int)(r % HD_);
    long bh = r / HD_;
    int  hh = (int)(bh % H_);
    int  b  = (int)(bh / H_);
    dst[r * SPK_ + s] = src[((long)b * S_ + s) * D_ + hh * HD_ + d];
  }
}

// ---------------------------------------------------------------------------
// softmax over key dim with scale + additive mask; fp32 scores (stride SP2)
// -> bf16 probs (stride SP2, cols [S,SP2) zero-filled so the context GEMM's
// padded K reads exact zeros). One block per (b,h,q) row; S=197 <= 256.
// ---------------------------------------------------------------------------
__global__ __launch_bounds__(256) void softmax_mask_k(
    const float* __restrict__ sc, const float* __restrict__ mask,
    __bf16* __restrict__ pr, int H, int S, float scale)
{
  __shared__ float red[256];
  long row = blockIdx.x;                    // b*H*S + h*S + q
  int b = (int)(row / ((long)H * S));
  const float* srow = sc + row * SP2_;
  const float* mrow = mask + (long)b * S;
  int tid = threadIdx.x;
  float v = -3.0e38f;
  if (tid < S) v = srow[tid] * scale + mrow[tid];
  red[tid] = v; __syncthreads();
  for (int st = 128; st > 0; st >>= 1) {
    if (tid < st) red[tid] = fmaxf(red[tid], red[tid + st]);
    __syncthreads();
  }
  float mx = red[0]; __syncthreads();
  float e = (tid < S) ? expf(v - mx) : 0.f;
  red[tid] = e; __syncthreads();
  for (int st = 128; st > 0; st >>= 1) {
    if (tid < st) red[tid] += red[tid + st];
    __syncthreads();
  }
  float inv = 1.f / red[0];
  if (tid < SP2_) pr[row * SP2_ + tid] = (tid < S) ? (__bf16)(e * inv)
                                                   : (__bf16)0.0f;
}

// token_score[b,k] = mean over (h,q) of probs[b,h,q,k]; block per (b,k)
__global__ __launch_bounds__(256) void token_score_k(
    const __bf16* __restrict__ pr, float* __restrict__ ts, int H, int S)
{
  __shared__ float red[256];
  int b = blockIdx.x / S, k = blockIdx.x % S;
  long base = ((long)b * H) * S * SP2_ + k;
  int tid = threadIdx.x;
  float s = 0.f;
  for (int t = tid; t < H * S; t += 256)      // t = h*S + q
    s += (float)pr[base + (long)t * SP2_];
  red[tid] = s; __syncthreads();
  for (int st = 128; st > 0; st >>= 1) {
    if (tid < st) red[tid] += red[tid + st];
    __syncthreads();
  }
  if (tid == 0) ts[blockIdx.x] = red[0] / (float)(H * S);
}

// rank-bucket token pruning (stable descending argsort rank), block per batch
__global__ __launch_bounds__(256) void prune_k(
    float* __restrict__ mask, const float* __restrict__ ts,
    const float* __restrict__ rmask, const float* __restrict__ pz, int S)
{
  __shared__ float sc[256], mk[256], red[256];
  int b = blockIdx.x, tid = threadIdx.x;
  float m = 0.f, bin = 0.f;
  if (tid < S) { m = mask[(long)b * S + tid]; bin = (m > -1.0f) ? 1.f : 0.f; }
  mk[tid] = m;
  sc[tid] = (tid < S) ? ts[(long)b * S + tid] * bin : 0.f;
  red[tid] = bin; __syncthreads();
  for (int st = 128; st > 0; st >>= 1) {
    if (tid < st) red[tid] += red[tid + st];
    __syncthreads();
  }
  float eff = red[0]; __syncthreads();
  if (tid < S) {
    float si = sc[tid]; int rank = 0;
    for (int j = 0; j < S; ++j) {
      float sj = sc[j];
      rank += ((sj > si) || (sj == si && j < tid)) ? 1 : 0;
    }
    float rb = fminf(fmaxf((float)rank / (eff + 1e-6f), 0.f), 1.f) * (float)NB_;
    int ib = (int)rb;
    if (ib < 0) ib = 0;
    float keep = (ib >= NB_) ? 0.f : rmask[ib];
    float nm = (keep == 0.f) ? -10000.f : m;
    mask[(long)b * S + tid] = (pz[0] > 0.f) ? nm : mk[tid];
  }
}

__global__ void f32_to_bf16_k(const float* __restrict__ s,
                              __bf16* __restrict__ d, long n)
{
  long i  = (long)blockIdx.x * blockDim.x + threadIdx.x;
  long st = (long)gridDim.x * blockDim.x;
  for (; i < n; i += st) d[i] = (__bf16)s[i];
}

// ---------------------------------------------------------------------------
extern "C" void kernel_launch(void* const* d_in, const int* in_sizes, int n_in,
                              void* d_out, int out_size, void* d_ws, size_t ws_size,
                              hipStream_t stream)
{
  (void)in_sizes; (void)n_in; (void)out_size; (void)ws_size;

  const float* hs    = (const float*)d_in[0];
  const float* amask = (const float*)d_in[1];
  const float* tz    = (const float*)d_in[2];
  const float* pz    = (const float*)d_in[3];
  const float* ln1w  = (const float*)d_in[4];
  const float* ln1b  = (const float*)d_in[5];
  const float* qw    = (const float*)d_in[6];
  const float* qbv   = (const float*)d_in[7];
  const float* kw    = (const float*)d_in[8];
  const float* kbv   = (const float*)d_in[9];
  const float* vw    = (const float*)d_in[10];
  const float* vbv   = (const float*)d_in[11];
  const float* ow    = (const float*)d_in[12];
  const float* obv   = (const float*)d_in[13];
  const float* ln2w  = (const float*)d_in[14];
  const float* ln2b  = (const float*)d_in[15];
  const float* f1w   = (const float*)d_in[16];
  const float* f1bv  = (const float*)d_in[17];
  const float* f2w   = (const float*)d_in[18];
  const float* f2bv  = (const float*)d_in[19];

  char* base = (char*)d_ws;
  size_t off = 0;
  auto take = [&](size_t bytes) -> void* {
    void* p = base + off;
    off = (off + bytes + 255) & ~(size_t)255;
    return p;
  };

  const size_t LDD = (size_t)L_ * D_ * D_;
  const size_t LDF = (size_t)L_ * D_ * FF_;

  __bf16* qwbf  = (__bf16*)take(LDD * 2);
  __bf16* kwbf  = (__bf16*)take(LDD * 2);
  __bf16* vwbf  = (__bf16*)take(LDD * 2);
  __bf16* owbf  = (__bf16*)take(LDD * 2);
  __bf16* f1wbf = (__bf16*)take(LDF * 2);
  __bf16* f2wbf = (__bf16*)take(LDF * 2);
  float*  h     = (float*) take((size_t)MROW_ * D_ * 4);   // row-padded
  __bf16* xbf   = (__bf16*)take((size_t)MROW_ * D_ * 2);
  __bf16* qbf   = (__bf16*)take((size_t)MROW_ * D_ * 2);
  __bf16* kbf   = (__bf16*)take((size_t)MROW_ * D_ * 2);
  __bf16* vbf   = (__bf16*)take((size_t)MROW_ * D_ * 2);
  __bf16* kbT   = (__bf16*)take(((size_t)B_ * H_ * HD_ * SPK_ + 256) * 2);
  float*  scores= (float*) take((size_t)B_ * H_ * S_ * SP2_ * 4);
  __bf16* probs = (__bf16*)take(((size_t)B_ * H_ * S_ + 64) * SP2_ * 2);
  __bf16* ctxbf = (__bf16*)take((size_t)MROW_ * D_ * 2);
  __bf16* gbf   = (__bf16*)take((size_t)MROW_ * FF_ * 2);
  float*  maskb = (float*) take((size_t)B_ * S_ * 4);
  float*  tsc   = (float*) take((size_t)B_ * S_ * 4);

  hipMemcpyAsync(h, hs, (size_t)MS_ * D_ * 4, hipMemcpyDeviceToDevice, stream);
  hipMemcpyAsync(maskb, amask, (size_t)B_ * S_ * 4, hipMemcpyDeviceToDevice, stream);

  // weights -> bf16 (170MB bf16: L2-resident across the whole pass)
  f32_to_bf16_k<<<2048, 256, 0, stream>>>(qw,  qwbf,  (long)LDD);
  f32_to_bf16_k<<<2048, 256, 0, stream>>>(kw,  kwbf,  (long)LDD);
  f32_to_bf16_k<<<2048, 256, 0, stream>>>(vw,  vwbf,  (long)LDD);
  f32_to_bf16_k<<<2048, 256, 0, stream>>>(ow,  owbf,  (long)LDD);
  f32_to_bf16_k<<<4096, 256, 0, stream>>>(f1w, f1wbf, (long)LDF);
  f32_to_bf16_k<<<4096, 256, 0, stream>>>(f2w, f2wbf, (long)LDF);

  const float scale = 0.125f;   // 1/sqrt(HD)
  const dim3 blk(256);
  const dim3 gD((D_  + 127) / 128, (MS_ + 63) / 64, 1);
  const dim3 gF((FF_ + 127) / 128, (MS_ + 63) / 64, 1);
  const dim3 gS((S_  + 127) / 128, (S_  + 63) / 64, B_ * H_);
  const dim3 gC((HD_ + 127) / 128, (S_  + 63) / 64, B_ * H_);

  int pc = 0;
  for (int i = 0; i < L_; ++i) {
    const __bf16* qwi  = qwbf  + (size_t)i * D_ * D_;
    const __bf16* kwi  = kwbf  + (size_t)i * D_ * D_;
    const __bf16* vwi  = vwbf  + (size_t)i * D_ * D_;
    const __bf16* owi  = owbf  + (size_t)i * D_ * D_;
    const __bf16* f1wi = f1wbf + (size_t)i * D_ * FF_;
    const __bf16* f2wi = f2wbf + (size_t)i * FF_ * D_;

    // --- attention ---
    layernorm_bf16_k<<<MS_, 256, 0, stream>>>(h, ln1w + (size_t)i * D_,
                                              ln1b + (size_t)i * D_, xbf, D_);
    wmma_gemm_bf16<true,false,false,true,false><<<gD, blk, 0, stream>>>(
        xbf, qwi, qbv + (size_t)i * D_, nullptr, nullptr, qbf,
        MS_, D_, D_, D_, D_, D_, 0, 0, 0, 0, 0, 0, 1);
    wmma_gemm_bf16<true,false,false,true,false><<<gD, blk, 0, stream>>>(
        xbf, kwi, kbv + (size_t)i * D_, nullptr, nullptr, kbf,
        MS_, D_, D_, D_, D_, D_, 0, 0, 0, 0, 0, 0, 1);
    wmma_gemm_bf16<true,false,false,true,false><<<gD, blk, 0, stream>>>(
        xbf, vwi, vbv + (size_t)i * D_, nullptr, nullptr, vbf,
        MS_, D_, D_, D_, D_, D_, 0, 0, 0, 0, 0, 0, 1);
    ktranspose_bf16_k<<<2048, 256, 0, stream>>>(kbf, kbT);

    // scores[b,h,q,k] = Q[b,q,h,:] . K^T[b,h,:,k]
    wmma_gemm_bf16<false,false,true,false,false><<<gS, blk, 0, stream>>>(
        qbf, kbT, nullptr, nullptr, scores, nullptr,
        S_, S_, HD_, D_, SPK_, SP2_,
        (long)S_ * D_, (long)HD_, (long)H_ * HD_ * SPK_, (long)HD_ * SPK_,
        (long)H_ * S_ * SP2_, (long)S_ * SP2_, H_);
    softmax_mask_k<<<B_ * H_ * S_, 256, 0, stream>>>(scores, maskb, probs,
                                                     H_, S_, scale);
    // ctx[b,q,h,:] = probs[b,h,q,:] @ V[b,:,h,:]   (K padded to 224, A pad=0)
    wmma_gemm_bf16<false,false,false,true,false><<<gC, blk, 0, stream>>>(
        probs, vbf, nullptr, nullptr, nullptr, ctxbf,
        S_, HD_, SP2_, SP2_, D_, D_,
        (long)H_ * S_ * SP2_, (long)S_ * SP2_, (long)S_ * D_, (long)HD_,
        (long)S_ * D_, (long)HD_, H_);
    // h = ctx @ o_w + o_b + h
    wmma_gemm_bf16<true,true,true,false,false><<<gD, blk, 0, stream>>>(
        ctxbf, owi, obv + (size_t)i * D_, h, h, nullptr,
        MS_, D_, D_, D_, D_, D_, 0, 0, 0, 0, 0, 0, 1);

    // --- FFN ---
    layernorm_bf16_k<<<MS_, 256, 0, stream>>>(h, ln2w + (size_t)i * D_,
                                              ln2b + (size_t)i * D_, xbf, D_);
    // gbf = gelu(LN2(h) @ f1_w + f1_b)   (fused epilogue)
    wmma_gemm_bf16<true,false,false,true,true><<<gF, blk, 0, stream>>>(
        xbf, f1wi, f1bv + (size_t)i * FF_, nullptr, nullptr, gbf,
        MS_, FF_, D_, D_, FF_, FF_, 0, 0, 0, 0, 0, 0, 1);
    // h = gbf @ f2_w + f2_b + h
    wmma_gemm_bf16<true,true,true,false,false><<<gD, blk, 0, stream>>>(
        gbf, f2wi, f2bv + (size_t)i * D_, h, h, nullptr,
        MS_, D_, FF_, FF_, D_, D_, 0, 0, 0, 0, 0, 0, 1);

    // --- token pruning (after full layer, per reference) ---
    if (i == 3 || i == 6 || i == 9) {
      token_score_k<<<B_ * S_, 256, 0, stream>>>(probs, tsc, H_, S_);
      prune_k<<<B_, 256, 0, stream>>>(maskb, tsc, tz + pc * NB_, pz + pc, S_);
      ++pc;
    }
  }

  hipMemcpyAsync(d_out, h, (size_t)MS_ * D_ * 4, hipMemcpyDeviceToDevice, stream);
}
